// PatternAwareAttention_80367428043060
// MI455X (gfx1250) — matrix operand order repaired and checked by reading, not compiled, not run on hardware
//
#include <hip/hip_runtime.h>

// ---------------------------------------------------------------------------
// PatternAwareAttention for MI455X (gfx1250, wave32, WMMA).
// Pipeline (all bf16 intermediates, f32 accumulate):
//   K0 cvt_weights : q_w/v_w/c_w f32 -> bf16            (L2-resident, 384KB)
//   K1 key_softmax : softmax(s_bank/sqrt(DK)) -> keyT[h,l][dk][n] bf16
//   K2 qv_kernel   : fused q&v 1x1 conv (WMMA) + ReLU + per-head softmax(q)
//                    -> q[b,h,l][n][dk] bf16, vT[b,h,l][dk][n] bf16
//   K3 kv_kernel   : KV = K^T V per (b,h,l)  (WMMA)     -> kvT[y][x] bf16
//   K4 qkv_kernel  : Q @ KV per (b,h,l)      (WMMA, KV staged in LDS via
//                    global_load_async_to_lds_b128, ASYNCcnt-tracked)
//   K5 outconv     : c conv (WMMA) + bias + ReLU        -> f32 out [B,C,N,L]
//
// Round-3 change: fix async-to-LDS builtin call — hipcc's diagnostic showed
// the parameter type is a pointer to v4i (int ext_vector_type(4)) with
// address-space qualifiers, not void*.
// ---------------------------------------------------------------------------

#define BB   32
#define CC   256
#define NN   1024
#define LL   12
#define HH   8
#define DKK  32
#define NLP  (NN * LL)      // 12288 positions per batch
#define MT   64             // position tile per workgroup
#define MTILES (NLP / MT)   // 192

typedef __bf16 v8bf  __attribute__((ext_vector_type(8)));
typedef __bf16 v16bf __attribute__((ext_vector_type(16)));
typedef float  v8f   __attribute__((ext_vector_type(8)));
typedef int    v4i   __attribute__((ext_vector_type(4)));

#if __has_builtin(__builtin_amdgcn_global_load_async_to_lds_b128)
#define HAS_ASYNC_LDS 1
typedef __attribute__((address_space(1))) v4i as1_v4i;   // global
typedef __attribute__((address_space(3))) v4i as3_v4i;   // LDS
#else
#define HAS_ASYNC_LDS 0
#endif

// Build one 16-lane-element fragment from two aligned 16B chunks.
static __device__ __forceinline__ v16bf ld16(const __bf16* p0, const __bf16* p1) {
  v8bf a = *(const v8bf*)p0;
  v8bf b = *(const v8bf*)p1;
  return __builtin_shufflevector(a, b, 0,1,2,3,4,5,6,7,8,9,10,11,12,13,14,15);
}

static __device__ __forceinline__ v8f bwmma(v16bf a, v16bf b, v8f c) {
  // D = A x B + C, f32 accumulate, 16x16x32 bf16
  return __builtin_amdgcn_wmma_f32_16x16x32_bf16(false, a, false, b, (short)0, c,
                                                 false, false);
}

static __device__ __forceinline__ unsigned pk(float a, float b) {
  union { __bf16 h[2]; unsigned u; } p;
  p.h[0] = (__bf16)a;
  p.h[1] = (__bf16)b;
  return p.u;
}

// ---------------------------------------------------------------------------
// K0: convert the three 256x256 weight matrices to bf16 (row-major kept).
// ---------------------------------------------------------------------------
__global__ __launch_bounds__(256) void cvt_weights(
    const float* __restrict__ qw, const float* __restrict__ vw,
    const float* __restrict__ cw, __bf16* __restrict__ wq,
    __bf16* __restrict__ wv, __bf16* __restrict__ wc) {
  const int i = blockIdx.x * 256 + threadIdx.x;   // 0 .. 65535
  wq[i] = (__bf16)qw[i];
  wv[i] = (__bf16)vw[i];
  wc[i] = (__bf16)cw[i];
}

// ---------------------------------------------------------------------------
// K1: key softmax.  s_bank [H,L,N,DK] f32 -> keyT [(h*L+l)][dk][n] bf16
// (transposed so the kv-GEMM A-fragment reads are K(n)-contiguous).
// ---------------------------------------------------------------------------
__global__ __launch_bounds__(256) void key_softmax(
    const float* __restrict__ sb, __bf16* __restrict__ keyT) {
  const int t  = blockIdx.x * 256 + threadIdx.x;  // (h*L + l)*N + n
  const int n  = t % NN;
  const int hl = t / NN;
  const float scale = 0.1767766952966369f;        // 1/sqrt(DK)
  const float* src = sb + (size_t)t * DKK;
  float v[DKK];
  #pragma unroll
  for (int i = 0; i < DKK; i += 4) {
    const float4 f = *(const float4*)(src + i);
    v[i] = f.x; v[i+1] = f.y; v[i+2] = f.z; v[i+3] = f.w;
  }
  float mx = -3.0e38f;
  #pragma unroll
  for (int i = 0; i < DKK; ++i) mx = fmaxf(mx, v[i]);
  float s = 0.f;
  #pragma unroll
  for (int i = 0; i < DKK; ++i) { v[i] = __expf((v[i] - mx) * scale); s += v[i]; }
  const float inv = 1.f / s;
  __bf16* dst = keyT + (size_t)hl * DKK * NN + n;
  #pragma unroll
  for (int i = 0; i < DKK; ++i) dst[(size_t)i * NN] = (__bf16)(v[i] * inv);
}

// ---------------------------------------------------------------------------
// K2: fused q & v 1x1 conv + ReLU + per-head softmax on q.
// Block = 256 threads = 8 waves; wave w owns head w (channels 32w..32w+31).
// X tile (256 cin x 64 pos) staged in LDS as bf16, layout [m][k] (k fast)
// so each B-fragment is two contiguous 16B ds loads.
// Outputs: q[b,h,l][n][dk] bf16 (softmaxed), vT[b,h,l][dk][n] bf16.
// ---------------------------------------------------------------------------
__global__ __launch_bounds__(256) void qv_kernel(
    const float* __restrict__ x,  const __bf16* __restrict__ wq,
    const float* __restrict__ qb, const __bf16* __restrict__ wv,
    const float* __restrict__ vb, __bf16* __restrict__ qout,
    __bf16* __restrict__ vT) {
  __shared__ __align__(16) __bf16 lx[MT][264];    // 33.8 KB, padded rows
  const int bat = blockIdx.x / MTILES;
  const int m0  = (blockIdx.x % MTILES) * MT;
  const int tid = threadIdx.x;

  // cooperative fill: f32 -> bf16, transpose into [m][k]
  #pragma unroll 4
  for (int i = 0; i < 16; ++i) {
    const int task = i * 256 + tid;               // 4096 float4 tasks
    const int k  = task >> 4;
    const int mq = (task & 15) << 2;
    const float4 f = *(const float4*)(x + ((size_t)bat * CC + k) * NLP + m0 + mq);
    lx[mq + 0][k] = (__bf16)f.x;
    lx[mq + 1][k] = (__bf16)f.y;
    lx[mq + 2][k] = (__bf16)f.z;
    lx[mq + 3][k] = (__bf16)f.w;
  }
  __syncthreads();

  const int lane = tid & 31;
  const int w    = tid >> 5;          // wave id == head id
  const int l16  = lane & 15;
  const int hi   = (lane >> 4) & 1;
  const int r0   = w * DKK;           // head base output channel
  const float scale = 0.1767766952966369f;

  float bq0[8], bq1[8], bv0[8], bv1[8];
  #pragma unroll
  for (int i = 0; i < 8; ++i) {
    bq0[i] = qb[r0 + hi * 8 + i];
    bq1[i] = qb[r0 + 16 + hi * 8 + i];
    bv0[i] = vb[r0 + hi * 8 + i];
    bv1[i] = vb[r0 + 16 + hi * 8 + i];
  }
  const __bf16* aq0r = wq + (size_t)(r0 + l16) * CC;
  const __bf16* aq1r = aq0r + 16 * CC;
  const __bf16* av0r = wv + (size_t)(r0 + l16) * CC;
  const __bf16* av1r = av0r + 16 * CC;

  #pragma unroll 1      // keep one copy: occupancy > unroll for HBM-bound code
  for (int ct = 0; ct < 4; ++ct) {
    v8f q0, q1, v0, v1;
    #pragma unroll
    for (int i = 0; i < 8; ++i) { q0[i]=bq0[i]; q1[i]=bq1[i]; v0[i]=bv0[i]; v1[i]=bv1[i]; }

    #pragma unroll 2    // bounded unroll: pipeline loads vs WMMA, cap VGPRs
    for (int kt = 0; kt < 8; ++kt) {
      const int kb = kt * 32 + hi * 8;
      const __bf16* bp = &lx[ct * 16 + l16][kt * 32 + hi * 8];
      const v16bf bm = ld16(bp, bp + 16);                       // shared B frag
      q0 = bwmma(ld16(aq0r + kb, aq0r + kb + 16), bm, q0);
      q1 = bwmma(ld16(aq1r + kb, aq1r + kb + 16), bm, q1);
      v0 = bwmma(ld16(av0r + kb, av0r + kb + 16), bm, v0);
      v1 = bwmma(ld16(av1r + kb, av1r + kb + 16), bm, v1);
    }
    #pragma unroll
    for (int i = 0; i < 8; ++i) {
      q0[i] = fmaxf(q0[i], 0.f); q1[i] = fmaxf(q1[i], 0.f);
      v0[i] = fmaxf(v0[i], 0.f); v1[i] = fmaxf(v1[i], 0.f);
    }
    // softmax across the 32 head channels at each position (one shfl_xor 16)
    float mx = -3.0e38f;
    #pragma unroll
    for (int i = 0; i < 8; ++i) mx = fmaxf(mx, fmaxf(q0[i], q1[i]));
    mx = fmaxf(mx, __shfl_xor(mx, 16));
    float e0[8], e1[8], s = 0.f;
    #pragma unroll
    for (int i = 0; i < 8; ++i) {
      e0[i] = __expf((q0[i] - mx) * scale); s += e0[i];
      e1[i] = __expf((q1[i] - mx) * scale); s += e1[i];
    }
    s += __shfl_xor(s, 16);
    const float inv = 1.f / s;

    const int p = m0 + ct * 16 + l16;
    const int n = p / LL, l = p % LL;
    __bf16* qdst = qout + ((((size_t)bat * HH + w) * LL + l) * NN + n) * DKK;
    #pragma unroll
    for (int j = 0; j < 4; ++j) {
      *(unsigned*)(qdst + hi * 8 + 2 * j)      = pk(e0[2*j] * inv, e0[2*j+1] * inv);
      *(unsigned*)(qdst + 16 + hi * 8 + 2 * j) = pk(e1[2*j] * inv, e1[2*j+1] * inv);
    }
    __bf16* vdst = vT + (((size_t)bat * HH + w) * LL + l) * (size_t)DKK * NN + n;
    #pragma unroll
    for (int i = 0; i < 8; ++i) {
      vdst[(size_t)(hi * 8 + i) * NN]      = (__bf16)v0[i];
      vdst[(size_t)(16 + hi * 8 + i) * NN] = (__bf16)v1[i];
    }
  }
}

// ---------------------------------------------------------------------------
// K3: KV[x,y] = sum_n key[n,x] * v[n,y] per (b,h,l).  M=N'=32, K=1024.
// 2 waves/block: wave w does x rows [16w,16w+16), both y tiles.
// Output kvT[(b,h,l)][y][x] bf16 (transposed for K4's B fragments).
// ---------------------------------------------------------------------------
__global__ __launch_bounds__(64) void kv_kernel(
    const __bf16* __restrict__ keyT, const __bf16* __restrict__ vT,
    __bf16* __restrict__ kvT) {
  const int bhl = blockIdx.x;
  const int hl  = bhl % (HH * LL);
  const int lane = threadIdx.x & 31, w = threadIdx.x >> 5;
  const int l16 = lane & 15, hi = (lane >> 4) & 1;
  const int x0  = w * 16;

  const __bf16* arow  = keyT + ((size_t)hl * DKK + x0 + l16) * NN;
  const __bf16* b0row = vT + ((size_t)bhl * DKK + l16) * NN;
  const __bf16* b1row = b0row + (size_t)16 * NN;

  v8f acc0 = {}, acc1 = {};
  #pragma unroll 4
  for (int kt = 0; kt < 32; ++kt) {
    const int kb = kt * 32 + hi * 8;
    const int ko = kt * 32 + hi * 8;
    const v16bf a  = ld16(arow + kb, arow + kb + 16);
    acc0 = bwmma(a, ld16(b0row + ko, b0row + ko + 16), acc0);
    acc1 = bwmma(a, ld16(b1row + ko, b1row + ko + 16), acc1);
  }
  const size_t base = (size_t)bhl * DKK * DKK;
  #pragma unroll
  for (int yt = 0; yt < 2; ++yt) {
    const v8f acc = yt ? acc1 : acc0;
    const size_t row = base + (size_t)(yt * 16 + l16) * DKK + x0 + hi * 8;
    #pragma unroll
    for (int j = 0; j < 4; ++j)
      *(unsigned*)(kvT + row + 2 * j) = pk(acc[2*j], acc[2*j+1]);
  }
}

// ---------------------------------------------------------------------------
// K4: Out[n,y] = Q[n,:] @ KV[:,y] per (b,h,l).  K = DK = 32 -> 1 WMMA/tile.
// The 2KB KV tile is staged into LDS (async-to-LDS, ASYNCcnt-tracked) and
// reused by all 8 waves.  Output qkvT[b][m][c] bf16 (c fast) so K5 needs no
// LDS staging.
// ---------------------------------------------------------------------------
__global__ __launch_bounds__(256) void qkv_kernel(
    const __bf16* __restrict__ qbf, const __bf16* __restrict__ kvT,
    __bf16* __restrict__ qkvT) {
  __shared__ __align__(16) __bf16 skv[DKK * DKK];   // 2 KB KV tile
  const int bhl = blockIdx.x;
  const int l   = bhl % LL;
  const int bh  = bhl / LL;
  const int h   = bh % HH;
  const int bat = bh / HH;
  const int tid = threadIdx.x;
  const int lane = tid & 31, w = tid >> 5;
  const int l16 = lane & 15, hi = (lane >> 4) & 1;
  const size_t qbase  = (size_t)bhl * NN * DKK;
  const size_t kvbase = (size_t)bhl * DKK * DKK;

#if HAS_ASYNC_LDS
  if (tid < 128) {   // 128 x 16B = 2048B, ASYNCcnt-tracked DMA into LDS
    __builtin_amdgcn_global_load_async_to_lds_b128(
        (as1_v4i*)(kvT + kvbase + tid * 8), (as3_v4i*)&skv[tid * 8], 0, 0);
  }
#if __has_builtin(__builtin_amdgcn_s_wait_asynccnt)
  __builtin_amdgcn_s_wait_asynccnt(0);
#else
  asm volatile("s_wait_asynccnt 0x0" ::: "memory");
#endif
#else
  if (tid < 128) *(v8bf*)&skv[tid * 8] = *(const v8bf*)(kvT + kvbase + tid * 8);
#endif
  __syncthreads();

  #pragma unroll 2
  for (int it = 0; it < 8; ++it) {
    const int nb = w * 128 + it * 16;
    const __bf16* arow = qbf + qbase + (size_t)(nb + l16) * DKK + hi * 8;
    const v16bf a = ld16(arow, arow + 16);
    #pragma unroll
    for (int yt = 0; yt < 2; ++yt) {
      const __bf16* brow = skv + (size_t)(yt * 16 + l16) * DKK + hi * 8;
      v8f acc = {};
      acc = bwmma(a, ld16(brow, brow + 16), acc);
      const int c = h * DKK + yt * 16 + l16;
      #pragma unroll
      for (int i = 0; i < 8; ++i) {
        const int n = nb + hi * 8 + i;
        qkvT[((size_t)bat * NLP + (size_t)n * LL + l) * CC + c] = (__bf16)acc[i];
      }
    }
  }
}

// ---------------------------------------------------------------------------
// K5: output 1x1 conv + bias + ReLU -> f32 [B,C,N,L].
// X already bf16 with c contiguous per position -> B frags straight from L2.
// ---------------------------------------------------------------------------
__global__ __launch_bounds__(256) void outconv_kernel(
    const __bf16* __restrict__ xT, const __bf16* __restrict__ wc,
    const float* __restrict__ cb, float* __restrict__ out) {
  const int bat = blockIdx.x / MTILES;
  const int m0  = (blockIdx.x % MTILES) * MT;
  const int lane = threadIdx.x & 31, w = threadIdx.x >> 5;
  const int l16 = lane & 15, hi = (lane >> 4) & 1;
  __builtin_prefetch(xT + ((size_t)bat * NLP + m0) * CC, 0, 0);  // global_prefetch_b8

  #pragma unroll 1
  for (int rt = 0; rt < 2; ++rt) {
    const int r0 = w * 32 + rt * 16;
    const __bf16* wrow = wc + (size_t)(r0 + l16) * CC;
    float bias[8];
    #pragma unroll
    for (int i = 0; i < 8; ++i) bias[i] = cb[r0 + hi * 8 + i];

    #pragma unroll 1
    for (int ct = 0; ct < 4; ++ct) {
      v8f acc;
      #pragma unroll
      for (int i = 0; i < 8; ++i) acc[i] = bias[i];
      const __bf16* xrow = xT + ((size_t)bat * NLP + m0 + ct * 16 + l16) * CC;
      #pragma unroll 2
      for (int kt = 0; kt < 8; ++kt) {
        const int kb = kt * 32 + hi * 8;
        const __bf16* bp = xrow + kt * 32 + hi * 8;
        acc = bwmma(ld16(wrow + kb, wrow + kb + 16), ld16(bp, bp + 16), acc);
      }
      const int m = m0 + ct * 16 + l16;
      #pragma unroll
      for (int i = 0; i < 8; ++i)
        out[((size_t)bat * CC + r0 + hi * 8 + i) * NLP + m] = fmaxf(acc[i], 0.f);
    }
  }
}

// ---------------------------------------------------------------------------
extern "C" void kernel_launch(void* const* d_in, const int* in_sizes, int n_in,
                              void* d_out, int out_size, void* d_ws, size_t ws_size,
                              hipStream_t stream) {
  (void)in_sizes; (void)n_in; (void)out_size; (void)ws_size;
  const float* x  = (const float*)d_in[0];
  const float* qw = (const float*)d_in[1];
  const float* qb = (const float*)d_in[2];
  const float* vw = (const float*)d_in[3];
  const float* vb = (const float*)d_in[4];
  const float* cw = (const float*)d_in[5];
  const float* cb = (const float*)d_in[6];
  const float* sb = (const float*)d_in[7];

  char* ws = (char*)d_ws;
  size_t off = 0;
  __bf16* wqb  = (__bf16*)(ws + off); off += (size_t)CC * CC * 2;
  __bf16* wvb  = (__bf16*)(ws + off); off += (size_t)CC * CC * 2;
  __bf16* wcb  = (__bf16*)(ws + off); off += (size_t)CC * CC * 2;
  __bf16* keyT = (__bf16*)(ws + off); off += (size_t)HH * LL * DKK * NN * 2;
  __bf16* qbf  = (__bf16*)(ws + off); off += (size_t)BB * HH * LL * NN * DKK * 2;
  __bf16* vT   = (__bf16*)(ws + off); off += (size_t)BB * HH * LL * NN * DKK * 2;
  __bf16* kvT  = (__bf16*)(ws + off); off += (size_t)BB * HH * LL * DKK * DKK * 2;
  __bf16* qkvT = (__bf16*)(ws + off); off += (size_t)BB * NLP * CC * 2;

  cvt_weights   <<<(CC * CC) / 256, 256, 0, stream>>>(qw, vw, cw, wqb, wvb, wcb);
  key_softmax   <<<(HH * LL * NN) / 256, 256, 0, stream>>>(sb, keyT);
  qv_kernel     <<<BB * MTILES, 256, 0, stream>>>(x, wqb, qb, wvb, vb, qbf, vT);
  kv_kernel     <<<BB * HH * LL, 64, 0, stream>>>(keyT, vT, kvT);
  qkv_kernel    <<<BB * HH * LL, 256, 0, stream>>>(qbf, kvT, qkvT);
  outconv_kernel<<<BB * MTILES, 256, 0, stream>>>(qkvT, wcb, cb, (float*)d_out);
}
